// ConceptNet_21835613733374
// MI455X (gfx1250) — matrix operand order repaired and checked by reading, not compiled, not run on hardware
//
#include <hip/hip_runtime.h>
#include <hip/hip_bf16.h>

// ---------- types ----------
typedef __attribute__((ext_vector_type(16))) __bf16 v16bf;
typedef __attribute__((ext_vector_type(8)))  float  v8f;
typedef __attribute__((ext_vector_type(4)))  float  v4f;

// ---------- problem constants ----------
#define DIM        512
#define NCPT       64
#define NBANK      200000
#define BS         2048
#define NCLS       100
#define TOPK_K     64

// output layout (floats): orig_pred, y_pred, L1, L2, norm, concept_pred
#define OUT_ORIG   0
#define OUT_Y      (BS*NCLS)             // 204800
#define OUT_L1     (2*BS*NCLS)           // 409600
#define OUT_L2     (OUT_L1+1)
#define OUT_NM     (OUT_L1+2)
#define OUT_CP     (OUT_L1+3)            // 409603

// workspace byte offsets (all 256-aligned)
#define WS_GRAM    0                     // 64*64 f32      = 16384 B
#define WS_INV     16384                 // 64*64 f32
#define WS_CTW     32768                 // 64*100 f32     = 25600 B
#define WS_GM      58368                 // 64*100 f32
#define WS_CDOTS   83968                 // 64 f32
#define WS_BPACK   84224                 // 32768 bf16     = 65536 B
#define WS_BSQ     149760                // 200000 f32     = 800000 B
#define WS_D2      949760                // 64*200000 f32  = 51.2 MB (L2-resident)

// =====================================================================
// Small fp32 GEMMs (plain + transposed-A), one thread per output elem.
// =====================================================================
__global__ void gemm_nn(const float* __restrict__ A, const float* __restrict__ B,
                        float* __restrict__ C, int M, int N, int K) {
    int id = blockIdx.x * blockDim.x + threadIdx.x;
    if (id >= M * N) return;
    int m = id / N, n = id % N;
    const float* a = A + (size_t)m * K;
    float acc = 0.0f;
    for (int k = 0; k < K; ++k) acc = fmaf(a[k], B[(size_t)k * N + n], acc);
    C[id] = acc;
}

// out(M,N) = A^T(M,K) * B(K,N), A stored K x M row-major
__global__ void gemm_tn(const float* __restrict__ A, const float* __restrict__ B,
                        float* __restrict__ C, int M, int N, int K) {
    int id = blockIdx.x * blockDim.x + threadIdx.x;
    if (id >= M * N) return;
    int m = id / N, n = id % N;
    float acc = 0.0f;
    for (int k = 0; k < K; ++k)
        acc = fmaf(A[(size_t)k * M + m], B[(size_t)k * N + n], acc);
    C[id] = acc;
}

// =====================================================================
// 64x64 SPD Gauss-Jordan inverse, single block of 256 threads.
// =====================================================================
__global__ void invert64(const float* __restrict__ G, float* __restrict__ Inv) {
    __shared__ float M[64 * 64];
    __shared__ float I[64 * 64];
    __shared__ float colk[64];
    int tid = threadIdx.x;
    for (int i = tid; i < 4096; i += 256) {
        M[i] = G[i];
        I[i] = ((i >> 6) == (i & 63)) ? 1.0f : 0.0f;
    }
    __syncthreads();
    for (int k = 0; k < 64; ++k) {
        if (tid < 64) colk[tid] = M[tid * 64 + k];
        __syncthreads();
        float inv_piv = 1.0f / colk[k];
        if (tid < 128) {
            if (tid < 64) M[k * 64 + tid] *= inv_piv;
            else          I[k * 64 + (tid - 64)] *= inv_piv;
        }
        __syncthreads();
        for (int idx = tid; idx < 4096; idx += 256) {
            int r = idx >> 6, c = idx & 63;
            if (r != k) {
                float f = colk[r];
                M[r * 64 + c] = fmaf(-f, M[k * 64 + c], M[r * 64 + c]);
                I[r * 64 + c] = fmaf(-f, I[k * 64 + c], I[r * 64 + c]);
            }
        }
        __syncthreads();
    }
    for (int i = tid; i < 4096; i += 256) Inv[i] = I[i];
}

// =====================================================================
// Pack concept (DIM x NCPT, fp32 row-major) into WMMA bf16 B-fragments.
// Frag (s,t): 32x16 (KxN) slice, lane l holds 16 contiguous bf16:
//   lanes 0-15:  K = s*32 + (e<8 ? e : 8+e),  N = t*16 + l
//   lanes 16-31: K = s*32 + 8 + (e<8 ? e : 8+e), N = t*16 + (l-16)
// Stored so lane l's 16 values are contiguous: Bp[((s*4+t)*32+l)*16+e]
// =====================================================================
__global__ void pack_concept(const float* __restrict__ concept, __bf16* __restrict__ Bp) {
    int id = blockIdx.x * 256 + threadIdx.x;
    if (id >= 16 * 4 * 32 * 16) return;
    int e = id & 15;
    int l = (id >> 4) & 31;
    int t = (id >> 9) & 3;
    int s = id >> 11;
    int K = s * 32 + ((l < 16) ? 0 : 8) + e + ((e < 8) ? 0 : 8);
    int n = t * 16 + (l & 15);
    Bp[id] = (__bf16)concept[K * NCPT + n];
}

// =====================================================================
// Main WMMA kernel: d2[n][j] = max(c_sq[n] + |bank_j|^2 - 2<bank_j,c_n>, 0)
// for all 64 concepts, plus bank_sq[j]. 4 waves/block, 16 bank rows per
// wave, K=512 in 16 steps of V_WMMA_F32_16X16X32_BF16; fp32 bank stream
// converted to bf16 in-register (single HBM read of the 410 MB bank).
// 200000 rows = 3125 blocks * 64 rows exactly (no tail, EXEC all 1s).
// Row sums of squares are redistributed across lanes with shfl so d2 is
// formed in-register -> selection passes need 1 load per element.
// =====================================================================
__global__ __launch_bounds__(128) void bankdot(const float* __restrict__ bank,
                                               const __bf16* __restrict__ Bp,
                                               const float* __restrict__ gram,
                                               float* __restrict__ d2mat,
                                               float* __restrict__ bank_sq) {
    int tid  = threadIdx.x;
    int lane = tid & 31;
    int w    = tid >> 5;
    int r0   = blockIdx.x * 64 + w * 16;        // first row of this wave's slab
    int r    = r0 + (lane & 15);                // row this lane loads
    int koff = (lane < 16) ? 0 : 8;             // A-frag K sub-offset per lane half
    const float* ap = bank + (size_t)r * DIM + koff;
    const v16bf* bpv = (const v16bf*)Bp;

    v8f acc[4];
    acc[0] = (v8f)(0.0f); acc[1] = (v8f)(0.0f);
    acc[2] = (v8f)(0.0f); acc[3] = (v8f)(0.0f);
    float sq = 0.0f;

    for (int s = 0; s < 16; ++s) {
        const float* p = ap + s * 32;
        v16bf a;
#pragma unroll
        for (int e = 0; e < 8; ++e) {
            float x = p[e];
            sq = fmaf(x, x, sq);
            a[e] = (__bf16)x;
        }
#pragma unroll
        for (int e = 0; e < 8; ++e) {
            float x = p[16 + e];
            sq = fmaf(x, x, sq);
            a[8 + e] = (__bf16)x;
        }
#pragma unroll
        for (int t = 0; t < 4; ++t) {
            v16bf b = bpv[(size_t)((s * 4 + t) * 32 + lane)];
            acc[t] = __builtin_amdgcn_wmma_f32_16x16x32_bf16(
                false, a, false, b, (short)0, acc[t], false, false);
        }
    }

    // bank_sq: lane L has half of row (r0 + L%16); pair with L^16.
    // After this, EVERY lane holds the full sum for row r0 + (lane&15).
    sq += __shfl_xor(sq, 16, 32);
    if (lane < 16) bank_sq[r] = sq;

    // Redistribute row sums to accumulator-tile rows:
    // D element v of this lane is row j = r0 + base + v, whose sum lives
    // in lane (base + v).
    int base = (lane < 16) ? 0 : 8;
    float rowsq[8];
#pragma unroll
    for (int v = 0; v < 8; ++v) rowsq[v] = __shfl(sq, base + v, 32);

    int jb = r0 + base;
#pragma unroll
    for (int t = 0; t < 4; ++t) {
        int n = t * 16 + (lane & 15);
        float csq = gram[n * 64 + n];
        v8f d2;
#pragma unroll
        for (int v = 0; v < 8; ++v)
            d2[v] = fmaxf(csq + rowsq[v] - 2.0f * acc[t][v], 0.0f);
        float* dptr = d2mat + (size_t)n * NBANK + jb;
        *(v4f*)(dptr)     = __builtin_shufflevector(d2, d2, 0, 1, 2, 3);
        *(v4f*)(dptr + 4) = __builtin_shufflevector(d2, d2, 4, 5, 6, 7);
    }
}

// =====================================================================
// Per-concept 64th-smallest d2 via 8-bit radix select (4 histogram
// passes on the monotone IEEE bit pattern of d2 >= 0), then one final
// sweep summing the selected dots (dot recovered from d2). 5 sweeps
// total vs 33 for binary search; d2 matrix is L2-resident.
// =====================================================================
__global__ __launch_bounds__(256) void select_topk(const float* __restrict__ d2mat,
                                                   const float* __restrict__ bank_sq,
                                                   const float* __restrict__ gram,
                                                   float* __restrict__ cdots) {
    int n   = blockIdx.x;
    int tid = threadIdx.x;
    const float csq = gram[n * 64 + n];
    const float* row = d2mat + (size_t)n * NBANK;

    __shared__ unsigned hist[256];
    __shared__ unsigned s_prefix;
    __shared__ int      s_kth;
    __shared__ float    s_f[3];   // sum_less, sum_eq, cnt_eq

    unsigned prefix = 0;
    int      kth    = TOPK_K;     // rank (1-based) within current prefix group
    const unsigned maskhi[4] = {0u, 0xFF000000u, 0xFFFF0000u, 0xFFFFFF00u};

#pragma unroll 1
    for (int pass = 0; pass < 4; ++pass) {
        int shift = 24 - 8 * pass;
        unsigned mh = maskhi[pass];
        hist[tid] = 0;
        __syncthreads();
        for (int j = tid; j < NBANK; j += 256) {
            unsigned b = __float_as_uint(row[j]);
            if ((b & mh) == (prefix & mh))
                atomicAdd(&hist[(b >> shift) & 0xFFu], 1u);
        }
        __syncthreads();
        if (tid == 0) {
            int acc = 0, i = 0;
            for (; i < 256; ++i) {
                acc += (int)hist[i];
                if (acc >= kth) break;
            }
            s_kth    = kth - (acc - (int)hist[i]);   // rank within chosen bin
            s_prefix = prefix | ((unsigned)i << shift);
        }
        __syncthreads();
        prefix = s_prefix;
        kth    = s_kth;
        __syncthreads();
    }
    unsigned T    = prefix;       // exact bit pattern of 64th-smallest d2
    int      need = kth;          // # of ties at T inside the top-64

    if (tid < 3) s_f[tid] = 0.0f;
    __syncthreads();
    float sl = 0.0f, se = 0.0f, ce = 0.0f;
    for (int j = tid; j < NBANK; j += 256) {
        float d2 = row[j];
        unsigned b = __float_as_uint(d2);
        if (b <= T) {
            float dot = 0.5f * (csq + bank_sq[j] - d2);
            if (b < T) sl += dot;
            else       { se += dot; ce += 1.0f; }
        }
    }
    atomicAdd(&s_f[0], sl);
    atomicAdd(&s_f[1], se);
    atomicAdd(&s_f[2], ce);
    __syncthreads();
    if (tid == 0) {
        float res = s_f[0] +
                    ((s_f[2] > 0.0f) ? s_f[1] * ((float)need / s_f[2]) : 0.0f);
        cdots[n] = res / (float)TOPK_K;
    }
}

// =====================================================================
// Scalars: L_sparse_1 = mean(cdots); L_sparse_2 = (sum(gram)-tr)/4096;
// norm_metrics = tr/4096.
// =====================================================================
__global__ __launch_bounds__(256) void finalize(const float* __restrict__ gram,
                                                const float* __restrict__ cdots,
                                                float* __restrict__ out) {
    __shared__ float s_g[256], s_t[256], s_d[256];
    int tid = threadIdx.x;
    float gs = 0.0f, tr = 0.0f;
    for (int i = tid; i < 4096; i += 256) {
        float v = gram[i];
        gs += v;
        if ((i >> 6) == (i & 63)) tr += v;
    }
    float ds = (tid < 64) ? cdots[tid] : 0.0f;
    s_g[tid] = gs; s_t[tid] = tr; s_d[tid] = ds;
    __syncthreads();
    for (int off = 128; off > 0; off >>= 1) {
        if (tid < off) {
            s_g[tid] += s_g[tid + off];
            s_t[tid] += s_t[tid + off];
            s_d[tid] += s_d[tid + off];
        }
        __syncthreads();
    }
    if (tid == 0) {
        out[OUT_L1] = s_d[0] / 64.0f;
        out[OUT_L2] = (s_g[0] - s_t[0]) / 4096.0f;
        out[OUT_NM] = s_t[0] / 4096.0f;
    }
}

// =====================================================================
// Host launcher
// =====================================================================
extern "C" void kernel_launch(void* const* d_in, const int* in_sizes, int n_in,
                              void* d_out, int out_size, void* d_ws, size_t ws_size,
                              hipStream_t stream) {
    const float* concept = (const float*)d_in[0];   // (512, 64)
    const float* embed   = (const float*)d_in[1];   // (2048, 512)
    const float* bank    = (const float*)d_in[2];   // (200000, 512)
    const float* w_head  = (const float*)d_in[3];   // (512, 100)
    (void)in_sizes; (void)n_in; (void)out_size; (void)ws_size;

    float* out = (float*)d_out;
    char*  ws  = (char*)d_ws;
    float*  gram    = (float*)(ws + WS_GRAM);
    float*  invg    = (float*)(ws + WS_INV);
    float*  ctw     = (float*)(ws + WS_CTW);
    float*  gm      = (float*)(ws + WS_GM);
    float*  cdots   = (float*)(ws + WS_CDOTS);
    __bf16* bpack   = (__bf16*)(ws + WS_BPACK);
    float*  bank_sq = (float*)(ws + WS_BSQ);
    float*  d2mat   = (float*)(ws + WS_D2);

    // 1) pack concept into bf16 WMMA B-fragments (64 KB, stays hot in L2)
    pack_concept<<<128, 256, 0, stream>>>(concept, bpack);

    // 2) gram = C^T C  (64x64, K=512)  -- diagonal doubles as c_sq
    gemm_tn<<<(NCPT * NCPT + 255) / 256, 256, 0, stream>>>(concept, concept, gram,
                                                           NCPT, NCPT, DIM);
    // 3) inv(gram)  (SPD Gauss-Jordan)
    invert64<<<1, 256, 0, stream>>>(gram, invg);

    // 4) CtW = C^T W  (64x100, K=512)
    gemm_tn<<<(NCPT * NCLS + 255) / 256, 256, 0, stream>>>(concept, w_head, ctw,
                                                           NCPT, NCLS, DIM);
    // 5) Gm = inv(gram) @ CtW  (64x100, K=64)
    gemm_nn<<<(NCPT * NCLS + 255) / 256, 256, 0, stream>>>(invg, ctw, gm,
                                                           NCPT, NCLS, NCPT);

    // 6) orig_pred = E @ W  (2048x100, K=512)
    gemm_nn<<<(BS * NCLS + 255) / 256, 256, 0, stream>>>(embed, w_head, out + OUT_ORIG,
                                                         BS, NCLS, DIM);
    // 7) concept_pred = E @ C  (2048x64, K=512)
    gemm_nn<<<(BS * NCPT + 255) / 256, 256, 0, stream>>>(embed, concept, out + OUT_CP,
                                                         BS, NCPT, DIM);
    // 8) y_pred = concept_pred @ Gm  (2048x100, K=64)
    gemm_nn<<<(BS * NCLS + 255) / 256, 256, 0, stream>>>(out + OUT_CP, gm, out + OUT_Y,
                                                         BS, NCLS, NCPT);

    // 9) heavy op: d2 = ||c||^2 + ||b||^2 - 2 bank@C via bf16 WMMA (one HBM pass)
    bankdot<<<NBANK / 64, 128, 0, stream>>>(bank, bpack, gram, d2mat, bank_sq);

    // 10) per-concept top-64 via 4-pass radix select + 1 sum sweep
    select_topk<<<NCPT, 256, 0, stream>>>(d2mat, bank_sq, gram, cdots);

    // 11) scalar outputs
    finalize<<<1, 256, 0, stream>>>(gram, cdots, out);
}
